// MultiHeadDotProductAttention_11038065951434
// MI455X (gfx1250) — compile-verified, matrix operand order checked by reading, not run on hardware
//
#include <hip/hip_runtime.h>
#include <hip/hip_bf16.h>

typedef __bf16 v8bf  __attribute__((ext_vector_type(8)));
typedef __bf16 v16bf __attribute__((ext_vector_type(16)));
typedef float  v8f   __attribute__((ext_vector_type(8)));

#define TT  2048
#define DD  1024
#define HH  16
#define HDD 64
#define BBB 2

#if defined(__has_builtin)
#  if __has_builtin(__builtin_amdgcn_s_wait_asynccnt)
#    define HAVE_WAIT_ASYNC 1
#  endif
#endif

#ifdef HAVE_WAIT_ASYNC
#  define WAIT_ASYNC(n) __builtin_amdgcn_s_wait_asynccnt(n)
#else
#  define WAIT_ASYNC(n) asm volatile("s_wait_asynccnt %0" ::"i"(n) : "memory")
#endif

__device__ __forceinline__ unsigned short f2bf(float f) {
  unsigned int u = __float_as_uint(f);
  u += 0x7FFFu + ((u >> 16) & 1u);   // round-to-nearest-even
  return (unsigned short)(u >> 16);
}

__device__ __forceinline__ v8bf ld8(const unsigned short* p) {
  return *(const v8bf*)p;
}

__device__ __forceinline__ v16bf mk16(v8bf lo, v8bf hi) {
  v16bf r;
#pragma unroll
  for (int i = 0; i < 8; ++i) { r[i] = lo[i]; r[i + 8] = hi[i]; }
  return r;
}

// async 16B global -> LDS copy (ASYNCcnt-tracked on CDNA5).
// VDST = per-lane LDS byte address; VADDR = 64-bit global address, SADDR=off.
__device__ __forceinline__ void async_cp16(const unsigned short* g, unsigned short* l) {
  unsigned lds_off = (unsigned)(unsigned long long)l;  // aperture: LDS address is the pointer's low 32 bits
  asm volatile("global_load_async_to_lds_b128 %0, %1, off"
               :: "v"(lds_off), "v"((unsigned long long)g)
               : "memory");
}

// ---------------- converts ----------------
__global__ void cvt_f32_bf16(const float* __restrict__ in,
                             unsigned short* __restrict__ out, int n) {
  int i = blockIdx.x * blockDim.x + threadIdx.x;
  if (i < n) out[i] = f2bf(in[i]);
}

// in: [DD][DD] row-major; out[n][k] = in[k][n]  (transposed bf16 weights)
__global__ void cvt_f32_bf16_tr(const float* __restrict__ in,
                                unsigned short* __restrict__ out) {
  int i = blockIdx.x * blockDim.x + threadIdx.x;
  int n = i & (DD - 1);
  int k = i >> 10;
  out[(size_t)n * DD + k] = f2bf(in[(size_t)k * DD + n]);
}

// ---------------- GEMM: 4 waves, 64x64 tile, LDS-staged B (double buffered) ----
// X: [B*T][DD] bf16, WT: [N][DD] bf16 (W transposed).
// mode 0: out[b][h][t][hd] bf16 (Q,K)   mode 1: out[b][h][hd][t] bf16 (V)
// mode 2: out[m][n] f32
__device__ __forceinline__ void stage_b(const unsigned short* __restrict__ WT,
                                        unsigned short* buf, int n0, int k0) {
  const int t = threadIdx.x;       // 0..127: 64 rows x two 16-element halves
  const int row = t >> 1;
  const int co  = (t & 1) * 16;
  const unsigned short* src = WT + (size_t)(n0 + row) * DD + k0 + co;
  unsigned short* dst = buf + row * 32 + co;
  async_cp16(src, dst);
  async_cp16(src + 8, dst + 8);
}

__global__ void __launch_bounds__(128) gemm64(const unsigned short* __restrict__ X,
                                              const unsigned short* __restrict__ WT,
                                              void* __restrict__ outp, int mode) {
  __shared__ unsigned short bstage[2 * 64 * 32];   // 8 KB, double buffered

  const int t = threadIdx.x;
  const int lane = t & 31;
  const int w = t >> 5;                 // wave 0..3
  const int m0 = blockIdx.x * 64 + w * 16;
  const int n0 = blockIdx.y * 64;
  const int mr = lane & 15;
  const int koff = (lane < 16) ? 0 : 8;
  const int hi8  = (lane < 16) ? 0 : 8;

  v8f acc[4];
#pragma unroll
  for (int j = 0; j < 4; ++j) acc[j] = (v8f){};

  const unsigned short* xrow = X + (size_t)(m0 + mr) * DD;

  stage_b(WT, bstage, n0, 0);           // prologue: buffer 0

  const int NK = DD / 32;               // 32 k-steps
  for (int kt = 0; kt < NK; ++kt) {
    const int k0 = kt * 32;
    unsigned short* buf = bstage + (kt & 1) * (64 * 32);
    if (kt + 1 < NK) {
      stage_b(WT, bstage + ((kt + 1) & 1) * (64 * 32), n0, k0 + 32);
      WAIT_ASYNC(2);                    // current buffer's 2 loads done; next in flight
    } else {
      WAIT_ASYNC(0);
    }
    __syncthreads();                    // all waves' chunks visible

    v16bf a = mk16(ld8(xrow + k0 + koff), ld8(xrow + k0 + 16 + koff));
#pragma unroll
    for (int j = 0; j < 4; ++j) {
      const unsigned short* brow = buf + (j * 16 + mr) * 32;
      v16bf b = mk16(ld8(brow + koff), ld8(brow + 16 + koff));
      acc[j] = __builtin_amdgcn_wmma_f32_16x16x32_bf16(
          false, a, false, b, (short)0, acc[j], false, false);
    }
    __syncthreads();                    // reads done before buffer reuse
  }

  if (mode == 2) {
    float* out = (float*)outp;
#pragma unroll
    for (int j = 0; j < 4; ++j)
#pragma unroll
      for (int r = 0; r < 8; ++r) {
        int m = m0 + r + hi8;
        int n = n0 + j * 16 + mr;
        out[(size_t)m * DD + n] = acc[j][r];
      }
  } else {
    unsigned short* out = (unsigned short*)outp;
#pragma unroll
    for (int j = 0; j < 4; ++j)
#pragma unroll
      for (int r = 0; r < 8; ++r) {
        int m = m0 + r + hi8;
        int n = n0 + j * 16 + mr;
        int bb = m >> 11, tt = m & (TT - 1);
        int h = n >> 6, hd = n & 63;
        size_t off = (mode == 0)
            ? ((((size_t)bb * HH + h) * TT + tt) * HDD + hd)
            : ((((size_t)bb * HH + h) * HDD + hd) * TT + tt);
        out[off] = f2bf(acc[j][r]);
      }
  }
}

// ---------------- flash attention ----------------
// Q,K: [B][H][T][HD] bf16; V: [B][H][HD][T] bf16; AO: [B*T][H*HD] bf16
__global__ void attn_fwd(const unsigned short* __restrict__ Q,
                         const unsigned short* __restrict__ K,
                         const unsigned short* __restrict__ V,
                         unsigned short* __restrict__ AO) {
  __shared__ unsigned short ptile[16 * 16];

  const int lane = threadIdx.x;
  const int m0 = blockIdx.x * 16;
  const int h  = blockIdx.y;
  const int b  = blockIdx.z;
  const int mr   = lane & 15;
  const int koff = (lane < 16) ? 0 : 8;
  const int hi8  = (lane < 16) ? 0 : 8;

  const size_t head = ((size_t)b * HH + h) * (size_t)TT * HDD;

  const unsigned short* qrow = Q + head + (size_t)(m0 + mr) * HDD;
  v16bf qa0 = mk16(ld8(qrow + koff),      ld8(qrow + 16 + koff));
  v16bf qa1 = mk16(ld8(qrow + 32 + koff), ld8(qrow + 48 + koff));

  float mrun[8], lrun[8];
  v8f o[4];
#pragma unroll
  for (int r = 0; r < 8; ++r) { mrun[r] = -3.0e38f; lrun[r] = 0.0f; }
#pragma unroll
  for (int i = 0; i < 4; ++i) o[i] = (v8f){};

  v8bf zero8;
#pragma unroll
  for (int i = 0; i < 8; ++i) zero8[i] = (__bf16)0.0f;

  const int jmax = m0 >> 4;
  for (int j = 0; j <= jmax; ++j) {
    // ---- S = Q K^T (16x16 scores, K-dim = HD = 64 -> 2 WMMAs)
    const unsigned short* krow = K + head + (size_t)(j * 16 + mr) * HDD;
    v16bf kb0 = mk16(ld8(krow + koff),      ld8(krow + 16 + koff));
    v16bf kb1 = mk16(ld8(krow + 32 + koff), ld8(krow + 48 + koff));
    v8f s = (v8f){};
    s = __builtin_amdgcn_wmma_f32_16x16x32_bf16(false, qa0, false, kb0, (short)0, s, false, false);
    s = __builtin_amdgcn_wmma_f32_16x16x32_bf16(false, qa1, false, kb1, (short)0, s, false, false);

    // ---- scale + causal mask + online softmax
    float p[8];
#pragma unroll
    for (int r = 0; r < 8; ++r) {
      float sv = s[r] * 0.125f;                 // 1/sqrt(64)
      int row = m0 + r + hi8;
      int col = j * 16 + mr;
      if (col > row) sv = -3.0e38f;
      p[r] = sv;
    }
#pragma unroll
    for (int r = 0; r < 8; ++r) {
      float v = p[r];
      v = fmaxf(v, __shfl_xor(v, 1, 32));
      v = fmaxf(v, __shfl_xor(v, 2, 32));
      v = fmaxf(v, __shfl_xor(v, 4, 32));
      v = fmaxf(v, __shfl_xor(v, 8, 32));
      float mnew = fmaxf(mrun[r], v);
      float corr = __expf(mrun[r] - mnew);
      mrun[r] = mnew;
      float pe = __expf(p[r] - mnew);
      float rs = pe;
      rs += __shfl_xor(rs, 1, 32);
      rs += __shfl_xor(rs, 2, 32);
      rs += __shfl_xor(rs, 4, 32);
      rs += __shfl_xor(rs, 8, 32);
      lrun[r] = lrun[r] * corr + rs;
#pragma unroll
      for (int i = 0; i < 4; ++i) o[i][r] *= corr;
      ptile[(r + hi8) * 16 + mr] = f2bf(pe);    // C-layout -> LDS [m][n]
    }
    __syncthreads();

    // ---- re-load P in A-fragment layout (K = 16 keys active, upper half 0)
    v16bf pa = mk16(ld8(&ptile[mr * 16 + koff]), zero8);

    // ---- O += P * V   (4 WMMAs cover HD = 64)
#pragma unroll
    for (int i = 0; i < 4; ++i) {
      const unsigned short* vrow = V + head + (size_t)(i * 16 + mr) * TT + j * 16 + koff;
      v16bf vb = mk16(ld8(vrow), zero8);
      o[i] = __builtin_amdgcn_wmma_f32_16x16x32_bf16(
          false, pa, false, vb, (short)0, o[i], false, false);
    }
    __syncthreads();
  }

  // ---- normalize + store AO[b*T + m][h*64 + hd]
#pragma unroll
  for (int i = 0; i < 4; ++i) {
#pragma unroll
    for (int r = 0; r < 8; ++r) {
      int m = m0 + r + hi8;
      int col = h * HDD + i * 16 + mr;
      AO[(size_t)(b * TT + m) * (HH * HDD) + col] = f2bf(o[i][r] / lrun[r]);
    }
  }
}

// ---------------- launch ----------------
extern "C" void kernel_launch(void* const* d_in, const int* in_sizes, int n_in,
                              void* d_out, int out_size, void* d_ws, size_t ws_size,
                              hipStream_t stream) {
  const float* Xq  = (const float*)d_in[0];
  const float* Xkv = (const float*)d_in[1];
  // d_in[2] = causal mask (ignored; causality hardcoded)
  const float* Wq  = (const float*)d_in[3];
  const float* Wk  = (const float*)d_in[4];
  const float* Wv  = (const float*)d_in[5];
  const float* Wo  = (const float*)d_in[6];
  float* out = (float*)d_out;

  unsigned short* ws = (unsigned short*)d_ws;
  const size_t NX = (size_t)BBB * TT * DD;   // 4M elems
  const size_t NW = (size_t)DD * DD;         // 1M elems
  unsigned short* XQb  = ws;
  unsigned short* XKVb = XQb + NX;
  unsigned short* WQt  = XKVb + NX;
  unsigned short* WKt  = WQt + NW;
  unsigned short* WVt  = WKt + NW;
  unsigned short* WOt  = WVt + NW;
  unsigned short* Qb   = WOt + NW;
  unsigned short* Kb   = Qb + NX;
  unsigned short* Vb   = Kb + NX;
  unsigned short* AOb  = Vb + NX;            // total 28M u16 = 56 MB

  cvt_f32_bf16<<<(int)(NX / 256), 256, 0, stream>>>(Xq,  XQb,  (int)NX);
  cvt_f32_bf16<<<(int)(NX / 256), 256, 0, stream>>>(Xkv, XKVb, (int)NX);
  cvt_f32_bf16_tr<<<(int)(NW / 256), 256, 0, stream>>>(Wq, WQt);
  cvt_f32_bf16_tr<<<(int)(NW / 256), 256, 0, stream>>>(Wk, WKt);
  cvt_f32_bf16_tr<<<(int)(NW / 256), 256, 0, stream>>>(Wv, WVt);
  cvt_f32_bf16_tr<<<(int)(NW / 256), 256, 0, stream>>>(Wo, WOt);

  dim3 g1(BBB * TT / 64, DD / 64);           // 64 x 16 blocks, 4 waves each
  gemm64<<<g1, 128, 0, stream>>>(XQb,  WQt, (void*)Qb, 0);
  gemm64<<<g1, 128, 0, stream>>>(XKVb, WKt, (void*)Kb, 0);
  gemm64<<<g1, 128, 0, stream>>>(XKVb, WVt, (void*)Vb, 1);

  attn_fwd<<<dim3(TT / 16, HH, BBB), 32, 0, stream>>>(Qb, Kb, Vb, AOb);

  gemm64<<<g1, 128, 0, stream>>>(AOb, WOt, (void*)out, 2);
}
